// BilinearSeqAttn_37718402794169
// MI455X (gfx1250) — compile-verified, matrix operand order checked by reading, not run on hardware
//
#include <hip/hip_runtime.h>

// Problem constants (from reference setup)
#define DHID 768
#define LQ   1024
#define LC   2048
#define NB   8
#define SCALEF 0.03608439182435161f   // 1/sqrt(768)
#define NEGINF -1e30f

typedef __attribute__((ext_vector_type(16))) __bf16 v16bf;
typedef __attribute__((ext_vector_type(8)))  float  v8f;

union BFrag { v16bf v; uint4 q[2]; unsigned short u[16]; };

// Scheduling fence: forbid the post-RA scheduler from sinking prefetch loads
// down to their consumers (which re-serializes the K-loop into load->wait0->wmma).
#if defined(__has_builtin)
#if __has_builtin(__builtin_amdgcn_sched_barrier)
#define SCHED_FENCE() __builtin_amdgcn_sched_barrier(0)
#endif
#endif
#ifndef SCHED_FENCE
#define SCHED_FENCE()
#endif

__device__ __forceinline__ unsigned short f2bf(float x) {
  unsigned int u = __float_as_uint(x);
  u += 0x7fffu + ((u >> 16) & 1u);          // round-to-nearest-even
  return (unsigned short)(u >> 16);
}

// Double-buffered K-loop GEMM core: one 16-row A strip (row base `arow`,
// per-lane layout offset already applied), NT B tiles with per-lane row
// pointers bb[t] (layout offset applied), K = NK*32. Iteration kk+1's
// fragments are issued (and fenced) before iteration kk's WMMA chain, so
// the waitcnt pass emits partial s_wait_loadcnt and VMEM latency overlaps
// the WMMA chain.
template <int NT, int NK>
__device__ __forceinline__ void gemm_db(v8f (&acc)[NT],
                                        const unsigned short* arow,
                                        const unsigned short* const (&bb)[NT]) {
  BFrag a0, a1, b0[NT], b1[NT];
  auto ldA = [&](BFrag& A, int kk) {
    A.q[0] = *(const uint4*)(arow + kk * 32);
    A.q[1] = *(const uint4*)(arow + kk * 32 + 16);
  };
  auto ldB = [&](BFrag (&Bv)[NT], int kk) {
#pragma unroll
    for (int t = 0; t < NT; ++t) {
      Bv[t].q[0] = *(const uint4*)(bb[t] + kk * 32);
      Bv[t].q[1] = *(const uint4*)(bb[t] + kk * 32 + 16);
    }
  };
  auto mm = [&](BFrag& A, BFrag (&Bv)[NT]) {
#pragma unroll
    for (int t = 0; t < NT; ++t)
      acc[t] = __builtin_amdgcn_wmma_f32_16x16x32_bf16(false, A.v, false, Bv[t].v,
                                                       (short)0, acc[t], false, false);
  };
#pragma unroll
  for (int t = 0; t < NT; ++t) acc[t] = (v8f)(0.f);
  ldA(a0, 0); ldB(b0, 0);
  for (int kk = 0; kk < NK; kk += 2) {      // NK is even for all users
    ldA(a1, kk + 1); ldB(b1, kk + 1);       // prefetch kk+1
    SCHED_FENCE();
    mm(a0, b0);                             // consume kk (partial waits only)
    SCHED_FENCE();
    if (kk + 2 < NK) { ldA(a0, kk + 2); ldB(b0, kk + 2); }  // prefetch kk+2
    SCHED_FENCE();
    mm(a1, b1);                             // consume kk+1
    SCHED_FENCE();
  }
}

// ---------------- kernel 1: W (f32) -> bf16 ----------------
__global__ void k_convW(const float* __restrict__ W, unsigned short* __restrict__ Wb) {
  int i = blockIdx.x * blockDim.x + threadIdx.x;   // grid sized exactly
  Wb[i] = f2bf(W[i]);
}

// ------- kernel 2: question_hiddens -> bf16 rowmajor + bf16 transposed -------
__global__ void k_prepQ(const float* __restrict__ qh,
                        unsigned short* __restrict__ qhb,   // [B][Lq][D]
                        unsigned short* __restrict__ qhT) { // [B][D][Lq]
  __shared__ unsigned short tile[32][33];
  int b = blockIdx.z, q0 = blockIdx.y * 32, d0 = blockIdx.x * 32;
  int tx = threadIdx.x, ty = threadIdx.y;
#pragma unroll
  for (int s = 0; s < 4; ++s) {
    int i = ty + s * 8;
    size_t idx = ((size_t)(b * LQ + q0 + i)) * DHID + d0 + tx;
    unsigned short us = f2bf(qh[idx]);
    qhb[idx] = us;
    tile[i][tx] = us;
  }
  __syncthreads();
#pragma unroll
  for (int s = 0; s < 4; ++s) {
    int i = ty + s * 8;
    qhT[((size_t)(b * DHID + d0 + i)) * LQ + q0 + tx] = tile[tx][i];
  }
}

// ---------------- kernel 3: query = Qh @ W^T + bias (bf16 out) ----------------
#define AP_STR 776   // padded LDS row stride (ushorts): 1552B = 388 dwords == 4 mod 64 banks
__global__ void __launch_bounds__(256) k_proj(const unsigned short* __restrict__ qhb,
                                              const unsigned short* __restrict__ Wb,
                                              const float* __restrict__ bias,
                                              unsigned short* __restrict__ qryb) {
  __shared__ unsigned short at[16 * AP_STR];
  int m0 = blockIdx.x * 16;               // row in [0, B*Lq)
  int tid = threadIdx.x;
  { // stage A tile (16 x 768 bf16) into LDS
    const uint4* src = (const uint4*)(qhb + (size_t)m0 * DHID);
    for (int i = tid; i < 16 * 96; i += 256) {
      int r = i / 96, c = i % 96;
      *(uint4*)(at + r * AP_STR + c * 8) = src[(size_t)r * 96 + c];
    }
  }
  __syncthreads();
  int wave = tid >> 5, lane = tid & 31;
  int hf = lane >> 4, ln = lane & 15;
  int n0 = wave * 96;                     // 8 waves x 6 tiles x 16 = 768 cols
  const unsigned short* bb[6];
#pragma unroll
  for (int t = 0; t < 6; ++t)
    bb[t] = Wb + (size_t)(n0 + t * 16 + ln) * DHID + hf * 8;
  v8f acc[6];
  gemm_db<6, DHID / 32>(acc, at + ln * AP_STR + hf * 8, bb);
#pragma unroll
  for (int t = 0; t < 6; ++t) {
    int e = n0 + t * 16 + ln;
    float bv = bias[e];
#pragma unroll
    for (int r = 0; r < 8; ++r) {
      int m = m0 + r + hf * 8;
      qryb[(size_t)m * DHID + e] = f2bf(acc[t][r] + bv);
    }
  }
}

// ---------------- kernel 4: fused scores + softmax + attn output ----------------
#define CTX_STR 776
#define S_STR   1028
#define P_STR   1032
#define SM_CTXO 0
#define SM_SO   (16 * CTX_STR * 2)                 // 24832
#define SM_PO   (SM_SO + 16 * S_STR * 4)           // 90624
#define SM_ROWO (SM_PO + 16 * P_STR * 2)           // 123648
#define SM_TOT  (SM_ROWO + 64)                     // 123712 bytes dynamic LDS

__global__ void __launch_bounds__(256) k_attn(const float* __restrict__ ctxg,
                                              const int* __restrict__ qmask,
                                              const unsigned short* __restrict__ qryb,
                                              const unsigned short* __restrict__ qhTb,
                                              float* __restrict__ out) {
  extern __shared__ char smem[];
  unsigned short* ctx = (unsigned short*)(smem + SM_CTXO);
  float*          S   = (float*)(smem + SM_SO);
  unsigned short* P   = (unsigned short*)(smem + SM_PO);
  float*          rowinv = (float*)(smem + SM_ROWO);

  int m0 = blockIdx.x * 16;               // row over B*Lc
  int b  = m0 / LC;
  int tid = threadIdx.x;

  // Stage context tile to LDS (bf16) and stream fp32 copy to out[:, 0:768]
  {
    const float4* src = (const float4*)(ctxg + (size_t)m0 * DHID);
    for (int i = tid; i < 16 * 192; i += 256) {
      int r = i / 192, c = i % 192;
      float4 v = src[(size_t)r * 192 + c];
      *(float4*)(out + (size_t)(m0 + r) * (2 * DHID) + c * 4) = v;
      unsigned short* dp = ctx + r * CTX_STR + c * 4;
      dp[0] = f2bf(v.x); dp[1] = f2bf(v.y); dp[2] = f2bf(v.z); dp[3] = f2bf(v.w);
    }
  }
  __syncthreads();

  int wave = tid >> 5, lane = tid & 31;
  int hf = lane >> 4, ln = lane & 15;

  // Phase 1: scores. Wave w covers q in [w*128, w*128+128): 8 N-tiles, K = 768.
  {
    int q0w = wave * 128;
    const unsigned short* bb[8];
#pragma unroll
    for (int t = 0; t < 8; ++t)
      bb[t] = qryb + ((size_t)(b * LQ + q0w + t * 16 + ln)) * DHID + hf * 8;
    v8f acc[8];
    gemm_db<8, DHID / 32>(acc, ctx + ln * CTX_STR + hf * 8, bb);
#pragma unroll
    for (int t = 0; t < 8; ++t) {
      int q = q0w + t * 16 + ln;
#pragma unroll
      for (int r = 0; r < 8; ++r) S[(r + hf * 8) * S_STR + q] = acc[t][r];
    }
  }
  __syncthreads();

  // Phase 2: masked softmax per row; P holds UNNORMALIZED exp (bf16), rowinv = 1/sum.
  // 16 threads per row (one 16-lane group); each owns 16 float4 chunks strided by 16 (c, c+16, ...).
  {
    int r = tid >> 4, c = tid & 15;
    float4 vals[16];
    float mx = NEGINF;
#pragma unroll
    for (int i = 0; i < 16; ++i) {
      int q4 = i * 16 + c;                               // float4 index in row
      float4 s = *(const float4*)(S + r * S_STR + q4 * 4);
      int4 mk = *(const int4*)(qmask + b * LQ + q4 * 4);
      float4 v;
      v.x = mk.x ? s.x * SCALEF : NEGINF;
      v.y = mk.y ? s.y * SCALEF : NEGINF;
      v.z = mk.z ? s.z * SCALEF : NEGINF;
      v.w = mk.w ? s.w * SCALEF : NEGINF;
      vals[i] = v;
      mx = fmaxf(mx, fmaxf(fmaxf(v.x, v.y), fmaxf(v.z, v.w)));
    }
#pragma unroll
    for (int off = 8; off >= 1; off >>= 1) mx = fmaxf(mx, __shfl_xor(mx, off, 16));
    float sum = 0.f;
#pragma unroll
    for (int i = 0; i < 16; ++i) {
      int q4 = i * 16 + c;
      float4 v = vals[i];
      float ex = __expf(v.x - mx), ey = __expf(v.y - mx);
      float ez = __expf(v.z - mx), ew = __expf(v.w - mx);
      sum += (ex + ey) + (ez + ew);
      ushort4 p;
      p.x = f2bf(ex); p.y = f2bf(ey); p.z = f2bf(ez); p.w = f2bf(ew);
      *(ushort4*)(P + r * P_STR + q4 * 4) = p;           // ds_store_b64
    }
#pragma unroll
    for (int off = 8; off >= 1; off >>= 1) sum += __shfl_xor(sum, off, 16);
    if (c == 0) rowinv[r] = 1.f / sum;
  }
  __syncthreads();

  // Phase 3: O = P(16x1024) @ Qh(1024 x d-slice); wave w covers d in [w*96, w*96+96).
  {
    int d0 = wave * 96;
    const unsigned short* bb[6];
#pragma unroll
    for (int t = 0; t < 6; ++t)
      bb[t] = qhTb + ((size_t)(b * DHID + d0 + t * 16 + ln)) * LQ + hf * 8;
    v8f acc[6];
    gemm_db<6, LQ / 32>(acc, P + ln * P_STR + hf * 8, bb);
#pragma unroll
    for (int t = 0; t < 6; ++t) {
      int d = d0 + t * 16 + ln;
#pragma unroll
      for (int r = 0; r < 8; ++r) {
        int M = r + hf * 8;
        out[(size_t)(m0 + M) * (2 * DHID) + DHID + d] = acc[t][r] * rowinv[M];
      }
    }
  }
}

// ---------------- host launcher ----------------
extern "C" void kernel_launch(void* const* d_in, const int* in_sizes, int n_in,
                              void* d_out, int out_size, void* d_ws, size_t ws_size,
                              hipStream_t stream) {
  const float* ctx   = (const float*)d_in[0];
  // d_in[1] = context_mask (unused by reference forward)
  const float* qh    = (const float*)d_in[2];
  const int*   qmask = (const int*)d_in[3];
  const float* W     = (const float*)d_in[4];
  const float* bias  = (const float*)d_in[5];
  float* out = (float*)d_out;

  // Workspace layout (bf16 intermediates), total ~37.1 MB
  char* ws = (char*)d_ws;
  unsigned short* qhb  = (unsigned short*)(ws);
  unsigned short* qhT  = (unsigned short*)(ws + (size_t)NB * LQ * DHID * 2);
  unsigned short* Wb   = (unsigned short*)(ws + (size_t)NB * LQ * DHID * 4);
  unsigned short* qryb = (unsigned short*)(ws + (size_t)NB * LQ * DHID * 4 + (size_t)DHID * DHID * 2);

  (void)hipFuncSetAttribute((const void*)k_attn,
                            hipFuncAttributeMaxDynamicSharedMemorySize, SM_TOT);

  k_convW<<<(DHID * DHID) / 256, 256, 0, stream>>>(W, Wb);
  dim3 gT(DHID / 32, LQ / 32, NB), bT(32, 8);
  k_prepQ<<<gT, bT, 0, stream>>>(qh, qhb, qhT);
  k_proj<<<(NB * LQ) / 16, 256, 0, stream>>>(qhb, Wb, bias, qryb);
  k_attn<<<(NB * LC) / 16, 256, SM_TOT, stream>>>(ctx, qmask, qryb, qhT, out);
}
